// RNN_39573828665775
// MI455X (gfx1250) — compile-verified, matrix operand order checked by reading, not run on hardware
//
#include <hip/hip_runtime.h>
#include <hip/hip_bf16.h>

// ---------------------------------------------------------------------------
// GRU (flax GRUCell w/ reset mask), B=128, T=256, D=512, H=1024, gfx1250.
//
// Phase 0 (parallel prep): z -> bf16; Wi/Wh -> fragment-ready packed bf16;
//                          h(0) -> reset-masked bf16 + fp32 copies.
// Phase 1 (parallel):      x_proj = z @ Wi, bf16 WMMA, async-to-LDS staging
//                          (global_load_async_to_lds_b128 + s_wait_asynccnt).
// Phase 2 (serial scan):   256 step kernels; hh = h @ Wh via WMMA straight
//                          from the L2-resident masked bf16 h buffer (no LDS,
//                          no barrier on the critical path); fused gates;
//                          epilogue emits next step's pre-masked h (bf16+f32).
// ---------------------------------------------------------------------------

typedef __attribute__((ext_vector_type(16))) __bf16 v16bf;
typedef __attribute__((ext_vector_type(8)))  float  v8f;
typedef unsigned int u32x4 __attribute__((ext_vector_type(4)));
typedef unsigned int u32x2 __attribute__((ext_vector_type(2)));
typedef float        f32x4 __attribute__((ext_vector_type(4)));

// 32-bit LDS-address-space pointer: correct VDST operand for async-to-LDS and
// makes the __shared__ object escape so its later ds_load reads stay live.
typedef __attribute__((address_space(3))) unsigned short lds_ushort;

union Frag { v16bf v; u32x4 q[2]; };

__device__ __forceinline__ unsigned short f32_to_bf16(float f) {
  union { float f; unsigned u; } x; x.f = f;
  unsigned r = x.u + 0x7FFFu + ((x.u >> 16) & 1u);   // round-to-nearest-even
  return (unsigned short)(r >> 16);
}
__device__ __forceinline__ float bf16_to_f32(unsigned short h) {
  union { unsigned u; float f; } x; x.u = ((unsigned)h) << 16; return x.f;
}
__device__ __forceinline__ float sigmoidf(float x) {
  return 1.0f / (1.0f + __expf(-x));
}
__device__ __forceinline__ void wait_async_lds() {
#if __has_builtin(__builtin_amdgcn_s_wait_asynccnt)
  __builtin_amdgcn_s_wait_asynccnt(0);
#else
  asm volatile("s_wait_asynccnt 0x0" ::: "memory");
#endif
}

// ---------------------------------------------------------------------------
// Elementwise fp32 -> bf16 (vectorized by 4)
// ---------------------------------------------------------------------------
__global__ __launch_bounds__(256) void cvt_bf16_kernel(
    const float* __restrict__ src, unsigned short* __restrict__ dst, int n4) {
  int i = blockIdx.x * 256 + threadIdx.x;
  if (i >= n4) return;
  f32x4 v = *(const f32x4*)&src[(size_t)i * 4];
  u32x2 p;
  p.x = (unsigned)f32_to_bf16(v.x) | ((unsigned)f32_to_bf16(v.y) << 16);
  p.y = (unsigned)f32_to_bf16(v.z) | ((unsigned)f32_to_bf16(v.w) << 16);
  *(u32x2*)&dst[(size_t)i * 4] = p;
}

// ---------------------------------------------------------------------------
// Initial masked hidden state: h0 = reset[:,0] ? 0 : prev_state  (bf16 + f32)
// ---------------------------------------------------------------------------
__global__ __launch_bounds__(256) void hinit_kernel(
    const float* __restrict__ prev, const unsigned char* __restrict__ reset,
    unsigned short* __restrict__ hb, float* __restrict__ hf) {
  int i = blockIdx.x * 256 + threadIdx.x;        // 128*1024
  if (i >= 131072) return;
  int b = i >> 10;
  float v = reset[b * 256] ? 0.0f : prev[i];
  hb[i] = f32_to_bf16(v);
  hf[i] = v;
}

// ---------------------------------------------------------------------------
// Pack W [KW, 3072] fp32 into fragment-ready bf16 (16-bit B-operand striping):
//   frag f = jt*(KW/32)+kc; out[f*512 + lane*16 + e]:
//     j = jt*16 + (lane&15), hh = lane>>4,
//     K = kc*32 + (e<8 ? 8*hh+e : 16+8*hh+(e-8))
// ---------------------------------------------------------------------------
__global__ __launch_bounds__(256) void pack_weight_kernel(
    const float* __restrict__ W, unsigned short* __restrict__ out, int KW) {
  int idx = blockIdx.x * 256 + threadIdx.x;
  if (idx >= KW * 3072) return;
  int e = idx & 15, lane = (idx >> 4) & 31, frag = idx >> 9;
  int kcCount = KW / 32;
  int kc = frag % kcCount, jt = frag / kcCount;
  int j = jt * 16 + (lane & 15), hh = lane >> 4;
  int K = kc * 32 + (e < 8 ? hh * 8 + e : 16 + hh * 8 + (e - 8));
  out[idx] = f32_to_bf16(W[(size_t)K * 3072 + j]);
}

// ---------------------------------------------------------------------------
// x_proj GEMM: [32768 x 512]bf16 @ packed Wi -> bf16 [32768 x 3072].
// A tile staged into LDS with global_load_async_to_lds_b128 (ASYNCcnt path).
// Block = 128 threads (4 waves): 16 rows x 64 cols.
// ---------------------------------------------------------------------------
__global__ __launch_bounds__(128) void xproj_kernel(
    const unsigned short* __restrict__ zb, const unsigned short* __restrict__ wiP,
    unsigned short* __restrict__ xp) {
  constexpr int KW = 512, STR = KW + 8;             // +8 ushorts: bank spread
  __shared__ __align__(16) unsigned short sa[16 * STR];
  const int tid = threadIdx.x;
  const int lane = tid & 31, wave = tid >> 5;
  const int rowBase = blockIdx.y * 16;
  const int col = blockIdx.x * 64 + wave * 16;      // 0..3071

  // Async-stage A tile: 16 rows x 512 bf16 = 1024 x 16B transfers.
  for (int g = tid; g < 1024; g += 128) {
    int r = g >> 6, c8 = (g & 63) * 8;              // 64 groups of 8 bf16 per row
    const unsigned short* src = zb + (size_t)(rowBase + r) * KW + c8;
    lds_ushort* dst = (lds_ushort*)&sa[r * STR + c8];   // 32-bit LDS address
    asm volatile("global_load_async_to_lds_b128 %0, %1, off"
                 :: "v"(dst), "v"(src) : "memory");
  }
  wait_async_lds();
  __syncthreads();

  v8f acc = {};
  const int half = lane >> 4, aRow = lane & 15, nIdx = lane & 15;
  const unsigned short* bbase =
      wiP + (size_t)(col >> 4) * (KW / 32) * 512 + (size_t)lane * 16;
  for (int kc = 0; kc < KW / 32; ++kc) {
    Frag a, b;
    int k0 = kc * 32 + 8 * half;
    a.q[0] = *(const u32x4*)&sa[aRow * STR + k0];
    a.q[1] = *(const u32x4*)&sa[aRow * STR + k0 + 16];
    b.q[0] = *(const u32x4*)(bbase + kc * 512);
    b.q[1] = *(const u32x4*)(bbase + kc * 512 + 8);
    acc = __builtin_amdgcn_wmma_f32_16x16x32_bf16(false, a.v, false, b.v,
                                                  (short)0, acc, false, false);
  }
#pragma unroll
  for (int r = 0; r < 8; ++r) {                     // C/D: M=8*half+r, N=lane&15
    int row = rowBase + half * 8 + r;
    xp[(size_t)row * 3072 + col + nIdx] = f32_to_bf16(acc[r]);
  }
}

// ---------------------------------------------------------------------------
// One GRU step. A-frags come straight from the masked bf16 h buffer (256 KB,
// L2-resident) -- no LDS staging, no barrier on the serial critical path.
// Epilogue writes ys, next-step pre-masked h (bf16 + f32), h_final at t=255.
// Grid (16,8), block = 4 waves; each wave: 16x16 tile x 3 gate accumulators.
// ---------------------------------------------------------------------------
__global__ __launch_bounds__(128) void gru_step_kernel(
    const unsigned short* __restrict__ hb,    // masked bf16 h(t)  [128,1024]
    const float* __restrict__ hf,             // masked fp32 h(t)
    unsigned short* __restrict__ hb_next, float* __restrict__ hf_next,
    const unsigned short* __restrict__ whP,
    const unsigned short* __restrict__ xp,    // bf16 [32768, 3072]
    const float* __restrict__ bh,
    const unsigned char* __restrict__ reset,  // bool [128, 256]
    float* __restrict__ ys,                   // fp32 [128, 256, 1024]
    float* __restrict__ hfinal, int t) {
  constexpr int H = 1024, KC = H / 32;
  const int tid = threadIdx.x, lane = tid & 31, wave = tid >> 5;
  const int rowBase = blockIdx.y * 16;
  const int col = blockIdx.x * 64 + wave * 16;     // 0..1023
  const int half = lane >> 4, aRow = lane & 15, nIdx = lane & 15;

  v8f aR = {}, aZ = {}, aN = {};
  const unsigned short* aBase = hb + (size_t)(rowBase + aRow) * H + 8 * half;
  const size_t laneOff = (size_t)lane * 16;
  const unsigned short* bR = whP + (size_t)((0 * H + col) >> 4) * KC * 512 + laneOff;
  const unsigned short* bZ = whP + (size_t)((1 * H + col) >> 4) * KC * 512 + laneOff;
  const unsigned short* bN = whP + (size_t)((2 * H + col) >> 4) * KC * 512 + laneOff;

  for (int kc = 0; kc < KC; ++kc) {
    Frag a, fr, fz, fn;
    a.q[0]  = *(const u32x4*)(aBase + kc * 32);
    a.q[1]  = *(const u32x4*)(aBase + kc * 32 + 16);
    fr.q[0] = *(const u32x4*)(bR + kc * 512);
    fr.q[1] = *(const u32x4*)(bR + kc * 512 + 8);
    fz.q[0] = *(const u32x4*)(bZ + kc * 512);
    fz.q[1] = *(const u32x4*)(bZ + kc * 512 + 8);
    fn.q[0] = *(const u32x4*)(bN + kc * 512);
    fn.q[1] = *(const u32x4*)(bN + kc * 512 + 8);
    aR = __builtin_amdgcn_wmma_f32_16x16x32_bf16(false, a.v, false, fr.v,
                                                 (short)0, aR, false, false);
    aZ = __builtin_amdgcn_wmma_f32_16x16x32_bf16(false, a.v, false, fz.v,
                                                 (short)0, aZ, false, false);
    aN = __builtin_amdgcn_wmma_f32_16x16x32_bf16(false, a.v, false, fn.v,
                                                 (short)0, aN, false, false);
  }

  // Fused gate epilogue (fp32)
  const int j = col + nIdx;
  const float bhr = bh[j], bhz = bh[1024 + j], bhn = bh[2048 + j];
#pragma unroll
  for (int r = 0; r < 8; ++r) {
    int b = rowBase + half * 8 + r;
    size_t xrow = ((size_t)b * 256 + (size_t)t) * 3072;
    float xr = bf16_to_f32(xp[xrow + j]);
    float xz = bf16_to_f32(xp[xrow + 1024 + j]);
    float xn = bf16_to_f32(xp[xrow + 2048 + j]);
    float rg = sigmoidf(xr + aR[r] + bhr);
    float ug = sigmoidf(xz + aZ[r] + bhz);
    float ng = tanhf(xn + rg * (aN[r] + bhn));
    float hv = hf[(size_t)b * H + j];               // already reset-masked
    float hnew = (1.0f - ug) * ng + ug * hv;
    __builtin_nontemporal_store(hnew, &ys[((size_t)b * 256 + (size_t)t) * H + j]);
    // Pre-mask for step t+1 (mask known now): off the next step's critical path.
    float mn = (t + 1 < 256 && reset[b * 256 + t + 1]) ? 0.0f : 1.0f;
    float hm = hnew * mn;
    hb_next[(size_t)b * H + j] = f32_to_bf16(hm);
    hf_next[(size_t)b * H + j] = hm;
    if (t == 255) hfinal[(size_t)b * H + j] = hnew;
  }
}

// ---------------------------------------------------------------------------
// Workspace layout (bytes):
//   zb   bf16 z          : 32768*512*2  =  33,554,432   @ 0
//   wiP  packed Wi bf16  : 512*3072*2   =   3,145,728   @ 33,554,432
//   whP  packed Wh bf16  : 1024*3072*2  =   6,291,456   @ 36,700,160
//   hb0/hb1 bf16 h       : 2*128*1024*2 =     524,288   @ 42,991,616
//   hf0/hf1 fp32 h       : 2*128*1024*4 =   1,048,576   @ 43,515,904
//   xp   bf16 x_proj     : 32768*3072*2 = 201,326,592   @ 44,564,480
//   total ~235 MB
// ---------------------------------------------------------------------------
extern "C" void kernel_launch(void* const* d_in, const int* in_sizes, int n_in,
                              void* d_out, int out_size, void* d_ws, size_t ws_size,
                              hipStream_t stream) {
  (void)in_sizes; (void)n_in; (void)out_size; (void)ws_size;
  const float* prev          = (const float*)d_in[0];          // [128,1024]
  const float* z             = (const float*)d_in[1];          // [128,256,512]
  const unsigned char* reset = (const unsigned char*)d_in[2];  // [128,256] bool
  const float* Wi            = (const float*)d_in[3];          // [512,3072]
  const float* Wh            = (const float*)d_in[4];          // [1024,3072]
  const float* bh            = (const float*)d_in[5];          // [3072]

  float* out    = (float*)d_out;
  float* hfinal = out;                   // [128*1024]
  float* ys     = out + 128 * 1024;      // [128,256,1024]

  char* ws = (char*)d_ws;
  unsigned short* zb  = (unsigned short*)(ws + 0);
  unsigned short* wiP = (unsigned short*)(ws + 33554432);
  unsigned short* whP = (unsigned short*)(ws + 36700160);
  unsigned short* hb0 = (unsigned short*)(ws + 42991616);
  unsigned short* hb1 = (unsigned short*)(ws + 43253760);
  float*          hf0 = (float*)(ws + 43515904);
  float*          hf1 = (float*)(ws + 44040192);
  unsigned short* xp  = (unsigned short*)(ws + 44564480);

  hipLaunchKernelGGL(cvt_bf16_kernel, dim3(16384), dim3(256), 0, stream, z, zb, 4194304);
  hipLaunchKernelGGL(pack_weight_kernel, dim3(6144),  dim3(256), 0, stream, Wi, wiP, 512);
  hipLaunchKernelGGL(pack_weight_kernel, dim3(12288), dim3(256), 0, stream, Wh, whP, 1024);
  hipLaunchKernelGGL(hinit_kernel, dim3(512), dim3(256), 0, stream, prev, reset, hb0, hf0);
  hipLaunchKernelGGL(xproj_kernel, dim3(48, 2048), dim3(128), 0, stream, zb, wiP, xp);

  for (int t = 0; t < 256; ++t) {
    const unsigned short* hbi = (t & 1) ? hb1 : hb0;
    const float*          hfi = (t & 1) ? hf1 : hf0;
    unsigned short*       hbo = (t & 1) ? hb0 : hb1;
    float*                hfo = (t & 1) ? hf0 : hf1;
    hipLaunchKernelGGL(gru_step_kernel, dim3(16, 8), dim3(128), 0, stream,
                       hbi, hfi, hbo, hfo, whP, xp, bh, reset, ys, hfinal, t);
  }
}
